// KAN_Layer_87849261072974
// MI455X (gfx1250) — compile-verified, matrix operand order checked by reading, not run on hardware
//
#include <hip/hip_runtime.h>
#include <hip/hip_bf16.h>

// ---- CDNA5 WMMA / TDM types ----
typedef __attribute__((ext_vector_type(16))) _Float16     v16h;
typedef __attribute__((ext_vector_type(8)))  float        v8f;
typedef __attribute__((ext_vector_type(4)))  unsigned int v4u;
typedef __attribute__((ext_vector_type(8)))  int          v8i;
typedef __attribute__((ext_vector_type(4)))  int          v4i;

#define DIM_IN 128
#define DIM    64
#define BSZ    32
#define HH     64
#define WW     64
#define NPIX   (HH*WW)                 // 4096
#define NTOK   (BSZ*NPIX)              // 131072
#define GRIDG  5
#define DEGK   3
#define NB     (GRIDG+DEGK)            // 8 basis funcs per input
#define KSPL   (DIM*NB)                // 512

__device__ __forceinline__ v8f wmma16(v16h a, v16h b, v8f c) {
    // D = A(16x32 f16) * B(32x16 f16) + C(16x16 f32)
    return __builtin_amdgcn_wmma_f32_16x16x32_f16(
        /*neg_a=*/false, a, /*neg_b=*/false, b,
        /*c_mod=*/(short)0, c, /*reuse_a=*/false, /*reuse_b=*/false);
}

// fragment slot -> K offset (shared by A and B packing/consumption)
__device__ __forceinline__ int frag_koff(int lane, int slot) {
    return ((slot & 8) ? 16 : 0) + ((lane & 16) ? 8 : 0) + (slot & 7);
}

// Issue a 2D TDM tile load: [tile_d1 rows x tile_d0 elems] of 4-byte data,
// row stride `stride0` elements, global tile base `gaddr`, into LDS at
// byte offset `lds_addr`. Descriptor built per CDNA5 ISA.
__device__ __forceinline__ void tdm_load_tile_f32(unsigned long long gaddr,
                                                  unsigned int lds_addr,
                                                  unsigned int tensor_d0,
                                                  unsigned int tensor_d1,
                                                  unsigned int tile_d0,
                                                  unsigned int tile_d1,
                                                  unsigned int stride0) {
    // Group 0: count=1 | lds_addr | global_addr[56:0] | type=2
    v4u g0;
    g0.x = 1u;                                   // count=1, user mode
    g0.y = lds_addr;                             // LDS byte address
    g0.z = (unsigned int)gaddr;                  // global_addr[31:0]
    g0.w = ((unsigned int)(gaddr >> 32) & 0x01FFFFFFu) | (2u << 30); // [56:32] | type=2
    // Group 1: wg_mask=0, data_size=2 (4B), dims/strides per bitfield layout
    v8i g1;
    g1[0] = (int)(2u << 16);                           // data_size = 4 bytes
    g1[1] = (int)((tensor_d0 & 0xFFFFu) << 16);        // tensor_dim0[15:0]
    g1[2] = (int)((tensor_d0 >> 16) | ((tensor_d1 & 0xFFFFu) << 16));
    g1[3] = (int)((tensor_d1 >> 16) | (tile_d0 << 16)); // tile_dim0
    g1[4] = (int)(tile_d1);                            // tile_dim1 (tile_dim2=0)
    g1[5] = (int)stride0;                              // tensor_dim0_stride[31:0]
    g1[6] = 0;                                         // stride0[47:32] | stride1 lo
    g1[7] = 0;
    v4i z4 = {0, 0, 0, 0};
#if __clang_major__ >= 23
    v8i z8 = {0, 0, 0, 0, 0, 0, 0, 0};
    __builtin_amdgcn_tensor_load_to_lds(g0, g1, z4, z4, z8, 0);
#else
    __builtin_amdgcn_tensor_load_to_lds(g0, g1, z4, z4, 0);
#endif
}

// Cubic B-spline basis on uniform extended grid over [-1,1], G=5, K=3.
__device__ __forceinline__ void basis8(float x, float* out) {
    const float h = 2.0f / GRIDG; // 0.4
    float B[11];
#pragma unroll
    for (int q = 0; q < 11; ++q) {
        float tq  = (q - 3) * h - 1.0f;
        float tq1 = (q - 2) * h - 1.0f;
        B[q] = (x >= tq && x < tq1) ? 1.0f : 0.0f;
    }
#pragma unroll
    for (int p = 1; p <= DEGK; ++p) {
        float inv = 1.0f / (p * h);
#pragma unroll
        for (int q = 0; q < 11; ++q) {
            if (q < 11 - p) {
                float tq   = (q - 3) * h - 1.0f;
                float tqp1 = (q + p - 2) * h - 1.0f;   // t_{q+p+1}
                B[q] = ((x - tq) * B[q] + (tqp1 - x) * B[q + 1]) * inv;
            }
        }
    }
#pragma unroll
    for (int j = 0; j < NB; ++j) out[j] = B[j];
}

// ---------------------------------------------------------------------------
// Pack all GEMM weights into fragment-ready f16 buffers.
// id layout: fi*512 + lane*16 + slot
// ---------------------------------------------------------------------------
__global__ void pack_weights(const float* __restrict__ reduce_w,
                             const float* __restrict__ scale_base,
                             const float* __restrict__ coef,
                             const float* __restrict__ scale_sp,
                             const float* __restrict__ f2_w,
                             const float* __restrict__ g_w,
                             _Float16* __restrict__ pA_red,
                             _Float16* __restrict__ pB_base,
                             _Float16* __restrict__ pB_sp,
                             _Float16* __restrict__ pB_f2,
                             _Float16* __restrict__ pB_g) {
    int id   = blockIdx.x * blockDim.x + threadIdx.x;   // 0..32767
    int slot = id & 15;
    int lane = (id >> 4) & 31;
    int fi   = id >> 9;
    int n16  = lane & 15;
    int koff = frag_koff(lane, slot);

    {   // spline B: fi = kt*4 + nt, kt 0..15, nt 0..3, Kdim = 512
        int kt = fi >> 2, nt = fi & 3;
        int k = 32 * kt + koff;
        int n = nt * 16 + n16;
        int i = k >> 3, j = k & 7;
        pB_sp[id] = (_Float16)(coef[(i * DIM + n) * NB + j] * scale_sp[i * DIM + n]);
    }
    if (fi < 16) {  // reduce A: fi = ot*4 + kt, Kdim = 128, M = 64
        int ot = fi >> 2, kt = fi & 3;
        int k = 32 * kt + koff;
        int mrow = ot * 16 + n16;
        pA_red[id] = (_Float16)(reduce_w[mrow * DIM_IN + k]);
    }
    if (fi < 8) {   // base / f2 / g B: fi = kt*4 + nt, Kdim = 64
        int kt = fi >> 2, nt = fi & 3;
        int k = 32 * kt + koff;
        int n = nt * 16 + n16;
        pB_base[id] = (_Float16)scale_base[k * DIM + n];
        pB_f2[id]   = (_Float16)f2_w[n * DIM + k];  // conv weight [O,I]
        pB_g[id]    = (_Float16)g_w[n * DIM + k];
    }
}

// ---------------------------------------------------------------------------
// K1: 1x1 reduce conv as per-image GEMM W[64x128] * X[128x4096] via WMMA.
// The X tile [128 ch x 64 px] is DMA'd into LDS by the Tensor Data Mover,
// then B-fragments are gathered from LDS. Result ("inp") goes to d_out.
// ---------------------------------------------------------------------------
__global__ void reduce_gemm(const float* __restrict__ x,
                            const float* __restrict__ reduce_b,
                            const _Float16* __restrict__ pA_red,
                            float* __restrict__ red) {
    __shared__ float xtile[DIM_IN][64];     // 32 KB TDM destination

    int wave = threadIdx.x >> 5, lane = threadIdx.x & 31;
    int b  = blockIdx.x >> 6;               // 32 images
    int pb = blockIdx.x & 63;               // 64 pixel-blocks of 64
    int pixel_base = pb * 64;

    if (wave == 0) {
        unsigned long long gaddr =
            (unsigned long long)(uintptr_t)(x + (size_t)b * DIM_IN * NPIX + pixel_base);
        unsigned int lds_addr = (unsigned int)(uintptr_t)&xtile[0][0];
        tdm_load_tile_f32(gaddr, lds_addr,
                          /*tensor_d0=*/NPIX, /*tensor_d1=*/DIM_IN,
                          /*tile_d0=*/64, /*tile_d1=*/DIM_IN,
                          /*stride0=*/NPIX);
        __builtin_amdgcn_s_wait_tensorcnt(0);
    }
    __syncthreads();

    int hi = (lane >> 4) & 1;
    // 16 wave-tiles: ot(4) x pt(4); 8 waves -> 2 tiles each
#pragma unroll
    for (int it = 0; it < 2; ++it) {
        int idx = wave * 2 + it;
        int ot = idx >> 2;
        int pt = idx & 3;
        int pl = pt * 16 + (lane & 15);     // local pixel 0..63

        v8f acc = {};
#pragma unroll
        for (int kt = 0; kt < 4; ++kt) {
            v16h a = *(const v16h*)(pA_red + ((ot * 4 + kt) * 512) + lane * 16);
            v16h bf;
#pragma unroll
            for (int s = 0; s < 16; ++s) {
                int k = 32 * kt + ((s & 8) ? 16 : 0) + 8 * hi + (s & 7);
                bf[s] = (_Float16)xtile[k][pl];
            }
            acc = wmma16(a, bf, acc);
        }
        int obase = ot * 16 + 8 * hi;
        int p = pixel_base + pl;
#pragma unroll
        for (int r = 0; r < 8; ++r) {
            int o = obase + r;
            red[((size_t)b * DIM + o) * NPIX + p] = acc[r] + reduce_b[o];
        }
    }
}

// ---------------------------------------------------------------------------
// K2: depthwise 7x7 conv + BN (inference). NCHW in, token-major (NHWC) out.
// ---------------------------------------------------------------------------
__global__ void dwconv_bn(const float* __restrict__ red,
                          const float* __restrict__ dw_w,
                          const float* __restrict__ dw_b,
                          const float* __restrict__ dw_g,
                          const float* __restrict__ dw_beta,
                          const float* __restrict__ dw_m,
                          const float* __restrict__ dw_v,
                          float* __restrict__ xd_tm) {
    int tid = threadIdx.x;
    int w   = tid & 63;
    int row = blockIdx.x * 4 + (tid >> 6);  // (b,c,h): 131072 rows
    int h = row & 63;
    int c = (row >> 6) & 63;
    int b = row >> 12;
    const float* src = red + ((size_t)b * DIM + c) * NPIX;
    if (h + 4 < HH) __builtin_prefetch(src + (h + 4) * WW + w, 0, 1);  // global_prefetch_b8

    float acc = dw_b[c];
#pragma unroll
    for (int dh = 0; dh < 7; ++dh) {
        int hh = h + dh - 3;
        if (hh < 0 || hh >= HH) continue;
        const float* rp = src + hh * WW;
#pragma unroll
        for (int dw = 0; dw < 7; ++dw) {
            int ww = w + dw - 3;
            if (ww < 0 || ww >= WW) continue;
            acc += rp[ww] * dw_w[c * 49 + dh * 7 + dw];
        }
    }
    float inv = rsqrtf(dw_v[c] + 1e-5f);
    float v = dw_g[c] * (acc - dw_m[c]) * inv + dw_beta[c];
    size_t t = ((size_t)b * HH + h) * WW + w;
    xd_tm[t * DIM + c] = v;
}

// ---------------------------------------------------------------------------
// K3: fused KAN (silu-base GEMM + spline GEMM) + f2 GEMM + gate + g GEMM + BN.
// One wave owns 16 tokens; 8 waves / 128 tokens per workgroup.
// ---------------------------------------------------------------------------
__global__ void kan_fused(const float* __restrict__ xd_tm,
                          const _Float16* __restrict__ pB_base,
                          const _Float16* __restrict__ pB_sp,
                          const _Float16* __restrict__ pB_f2,
                          const _Float16* __restrict__ pB_g,
                          const float* __restrict__ f2_b,
                          const float* __restrict__ kan_bias,
                          const float* __restrict__ g_b,
                          const float* __restrict__ g_g,
                          const float* __restrict__ g_beta,
                          const float* __restrict__ g_m,
                          const float* __restrict__ g_v,
                          float* __restrict__ y3_tm) {
    __shared__ _Float16 lds_y[8][16][DIM];   // 16 KB: gate output staged per wave

    int wave = threadIdx.x >> 5, lane = threadIdx.x & 31;
    int t0 = blockIdx.x * 128 + wave * 16;
    int m  = lane & 15;
    int hi = (lane >> 4) & 1;
    const float* xrow = xd_tm + (size_t)(t0 + m) * DIM;

    // This lane's channel blocks for A-fragments: ch = 16*bb + 8*hi + e
    float xv[4][8];
#pragma unroll
    for (int bb = 0; bb < 4; ++bb)
#pragma unroll
        for (int e = 0; e < 8; ++e) xv[bb][e] = xrow[16 * bb + 8 * hi + e];

    v8f acc1[4] = {{0}, {0}, {0}, {0}};   // x1 = base + spline (+bias later)
    v8f acc2[4] = {{0}, {0}, {0}, {0}};   // x2 = f2 (+bias later)

    // ---- base (silu) GEMM + f2 GEMM, K = 64 (2 WMMA steps each) ----
#pragma unroll
    for (int kt = 0; kt < 2; ++kt) {
        v16h ax, as;
#pragma unroll
        for (int s = 0; s < 16; ++s) {
            int bb = 2 * kt + ((s & 8) ? 1 : 0);
            float xx = xv[bb][s & 7];
            ax[s] = (_Float16)xx;
            as[s] = (_Float16)(xx / (1.0f + __expf(-xx)));   // silu
        }
#pragma unroll
        for (int nt = 0; nt < 4; ++nt) {
            v16h wb = *(const v16h*)(pB_base + ((kt * 4 + nt) * 512) + lane * 16);
            v16h wf = *(const v16h*)(pB_f2  + ((kt * 4 + nt) * 512) + lane * 16);
            acc1[nt] = wmma16(as, wb, acc1[nt]);
            acc2[nt] = wmma16(ax, wf, acc2[nt]);
        }
    }

    // ---- spline GEMM, K = 512 (16 WMMA steps); basis computed in-lane ----
    for (int kt = 0; kt < 16; ++kt) {
        // slot s<8 -> input i1 = 4kt+hi, j=s ; s>=8 -> input i2 = 4kt+2+hi, j=s-8
        float b1[NB], b2[NB];
        basis8(xrow[4 * kt + hi], b1);
        basis8(xrow[4 * kt + 2 + hi], b2);
        v16h af;
#pragma unroll
        for (int s = 0; s < 8; ++s)  af[s] = (_Float16)b1[s];
#pragma unroll
        for (int s = 8; s < 16; ++s) af[s] = (_Float16)b2[s - 8];
#pragma unroll
        for (int nt = 0; nt < 4; ++nt) {
            v16h ws = *(const v16h*)(pB_sp + ((kt * 4 + nt) * 512) + lane * 16);
            acc1[nt] = wmma16(af, ws, acc1[nt]);
        }
    }

    // ---- biases + elementwise gate; stage y as f16 in LDS ----
#pragma unroll
    for (int nt = 0; nt < 4; ++nt) {
        int n = nt * 16 + m;
        float kb = kan_bias[n], fb = f2_b[n];
#pragma unroll
        for (int r = 0; r < 8; ++r) {
            float x1 = acc1[nt][r] + kb;
            float x2 = acc2[nt][r] + fb;
            lds_y[wave][r + 8 * hi][n] = (_Float16)(x1 * x2);
        }
    }
    __syncthreads();

    // ---- g 1x1 GEMM on gated y (K = 64), then BN, store token-major ----
    v8f acc3[4] = {{0}, {0}, {0}, {0}};
#pragma unroll
    for (int kt = 0; kt < 2; ++kt) {
        v16h ay;
#pragma unroll
        for (int s = 0; s < 16; ++s) {
            int ch = 32 * kt + ((s & 8) ? 16 : 0) + 8 * hi + (s & 7);
            ay[s] = lds_y[wave][m][ch];
        }
#pragma unroll
        for (int nt = 0; nt < 4; ++nt) {
            v16h wg = *(const v16h*)(pB_g + ((kt * 4 + nt) * 512) + lane * 16);
            acc3[nt] = wmma16(ay, wg, acc3[nt]);
        }
    }
#pragma unroll
    for (int nt = 0; nt < 4; ++nt) {
        int n = nt * 16 + m;
        float inv = rsqrtf(g_v[n] + 1e-5f);
        float gg = g_g[n], gbt = g_beta[n], gm = g_m[n], gbi = g_b[n];
#pragma unroll
        for (int r = 0; r < 8; ++r) {
            size_t t = (size_t)(t0 + r + 8 * hi);
            float v = acc3[nt][r] + gbi;
            y3_tm[t * DIM + n] = gg * (v - gm) * inv + gbt;
        }
    }
}

// ---------------------------------------------------------------------------
// K4: depthwise 7x7 on token-major y3 + residual (red aliases d_out, in-place).
// ---------------------------------------------------------------------------
__global__ void dwconv2_res(const float* __restrict__ y3_tm,
                            const float* __restrict__ dw2_w,
                            const float* __restrict__ dw2_b,
                            float* __restrict__ out /* holds red, updated in place */) {
    int tid = threadIdx.x;
    int c = tid & 63;
    size_t pix = (size_t)blockIdx.x * 4 + (tid >> 6);  // 131072 pixels
    int w = pix & 63;
    int h = (pix >> 6) & 63;
    int b = (int)(pix >> 12);

    float acc = dw2_b[c];
#pragma unroll
    for (int dh = 0; dh < 7; ++dh) {
        int hh = h + dh - 3;
        if (hh < 0 || hh >= HH) continue;
#pragma unroll
        for (int dw = 0; dw < 7; ++dw) {
            int ww = w + dw - 3;
            if (ww < 0 || ww >= WW) continue;
            size_t t = ((size_t)b * HH + hh) * WW + ww;
            acc += y3_tm[t * DIM + c] * dw2_w[c * 49 + dh * 7 + dw];
        }
    }
    size_t oi = ((size_t)b * DIM + c) * NPIX + (size_t)h * WW + w;
    out[oi] = out[oi] + acc;   // inp + y
}

// ---------------------------------------------------------------------------
extern "C" void kernel_launch(void* const* d_in, const int* in_sizes, int n_in,
                              void* d_out, int out_size, void* d_ws, size_t ws_size,
                              hipStream_t stream) {
    const float* x          = (const float*)d_in[0];
    const float* reduce_w   = (const float*)d_in[1];
    const float* reduce_b   = (const float*)d_in[2];
    const float* dw_w       = (const float*)d_in[3];
    const float* dw_b       = (const float*)d_in[4];
    const float* dw_g       = (const float*)d_in[5];
    const float* dw_beta    = (const float*)d_in[6];
    const float* dw_m       = (const float*)d_in[7];
    const float* dw_v       = (const float*)d_in[8];
    const float* f2_w       = (const float*)d_in[9];
    const float* f2_b       = (const float*)d_in[10];
    const float* coef       = (const float*)d_in[11];
    const float* scale_base = (const float*)d_in[12];
    const float* scale_sp   = (const float*)d_in[13];
    const float* kan_bias   = (const float*)d_in[14];
    const float* g_w        = (const float*)d_in[15];
    const float* g_b        = (const float*)d_in[16];
    const float* g_g        = (const float*)d_in[17];
    const float* g_beta     = (const float*)d_in[18];
    const float* g_m        = (const float*)d_in[19];
    const float* g_v        = (const float*)d_in[20];
    const float* dw2_w      = (const float*)d_in[21];
    const float* dw2_b      = (const float*)d_in[22];
    (void)in_sizes; (void)n_in; (void)out_size; (void)ws_size;

    char* ws = (char*)d_ws;
    float* red   = (float*)d_out;                       // 32 MB, aliases output (residual)
    float* xd_tm = (float*)ws;                          // 32 MB token-major xd
    float* y3_tm = (float*)(ws + (32ull << 20));        // 32 MB token-major y3
    _Float16* pA_red  = (_Float16*)(ws + (64ull << 20));
    _Float16* pB_base = pA_red  + 16 * 512;             //  8192 halfs
    _Float16* pB_sp   = pB_base +  8 * 512;             //  4096 halfs
    _Float16* pB_f2   = pB_sp   + 64 * 512;             // 32768 halfs
    _Float16* pB_g    = pB_f2   +  8 * 512;             //  4096 halfs

    pack_weights<<<128, 256, 0, stream>>>(reduce_w, scale_base, coef, scale_sp,
                                          f2_w, g_w,
                                          pA_red, pB_base, pB_sp, pB_f2, pB_g);
    reduce_gemm<<<2048, 256, 0, stream>>>(x, reduce_b, pA_red, red);
    dwconv_bn<<<32768, 256, 0, stream>>>(red, dw_w, dw_b, dw_g, dw_beta, dw_m, dw_v, xd_tm);
    kan_fused<<<1024, 256, 0, stream>>>(xd_tm, pB_base, pB_sp, pB_f2, pB_g,
                                        f2_b, kan_bias, g_b, g_g, g_beta, g_m, g_v, y3_tm);
    dwconv2_res<<<32768, 256, 0, stream>>>(y3_tm, dw2_w, dw2_b, red);
}
